// SwitchTransformerEncoderLayer_8203387536077
// MI455X (gfx1250) — matrix operand order repaired and checked by reading, not compiled, not run on hardware
//
#include <hip/hip_runtime.h>

// ---------------- problem constants ----------------
constexpr int kB  = 8;
constexpr int kT  = 1024;
constexpr int kD  = 1024;
constexpr int kH  = 16;
constexpr int kHD = 64;
constexpr int kE  = 4096;
constexpr int kNH = 16;
constexpr int kS  = kB * kT;      // 8192 tokens
constexpr int kC  = 3;            // ceil(1.25*8192/4096)
constexpr int k3D = 3 * kD;       // 3072

typedef __bf16 bf16;
typedef __attribute__((ext_vector_type(16))) __bf16 v16bf;
typedef __attribute__((ext_vector_type(8)))  __bf16 v8bf;
typedef __attribute__((ext_vector_type(8)))  float  v8f;

#define WMMA_BF16(a, b, c) \
  __builtin_amdgcn_wmma_f32_16x16x32_bf16(false, (a), false, (b), (short)0, (c), false, false)

// ---------------- WMMA fragment loaders (CDNA5 wave32 layouts) ----------------
// All loads are 128-bit vector loads (16B-aligned by construction).
// A 16x32 (MxK) bf16: lanes 0-15 M=lane, K={0..7,16..23}; lanes 16-31 M=lane-16, K={8..15,24..31}
__device__ inline v16bf ld_a16(const bf16* __restrict__ p, int ld) {
  const int lane = threadIdx.x & 31;
  const int m  = lane & 15;
  const int kb = (lane >> 4) << 3;          // 0 or 8
  const bf16* r0 = p + (size_t)m * ld + kb;
  v8bf lo = *(const v8bf*)(r0);
  v8bf hi = *(const v8bf*)(r0 + 16);
  return __builtin_shufflevector(lo, hi, 0, 1, 2, 3, 4, 5, 6, 7,
                                 8, 9, 10, 11, 12, 13, 14, 15);
}

// B 32x16 (KxN) with B stored row-major [N,K] (NT): lane holds column n=lane%16,
// lanes 0-15 K={0..7,16..23}, lanes 16-31 K={8..15,24..31} (2x b128)
__device__ inline v16bf ld_b_nt(const bf16* __restrict__ p, int ld) {
  const int lane = threadIdx.x & 31;
  const int n  = lane & 15;
  const int kb = (lane >> 4) << 3;          // 0 or 8
  const bf16* r0 = p + (size_t)n * ld + kb;
  v8bf lo = *(const v8bf*)(r0);
  v8bf hi = *(const v8bf*)(r0 + 16);
  return __builtin_shufflevector(lo, hi, 0, 1, 2, 3, 4, 5, 6, 7,
                                 8, 9, 10, 11, 12, 13, 14, 15);
}

// ---------------- generic cast / fill ----------------
__global__ void cast_k(const float* __restrict__ in, bf16* __restrict__ out, size_t n) {
  size_t i  = (size_t)blockIdx.x * blockDim.x + threadIdx.x;
  size_t st = (size_t)gridDim.x * blockDim.x;
  for (; i < n; i += st) out[i] = (bf16)in[i];
}
__global__ void fill_i32_k(int* __restrict__ p, int v, int n) {
  int i = blockIdx.x * 256 + threadIdx.x;
  if (i < n) p[i] = v;
}
__global__ void fill_f32_k(float* __restrict__ p, float v, size_t n) {
  size_t i  = (size_t)blockIdx.x * blockDim.x + threadIdx.x;
  size_t st = (size_t)gridDim.x * blockDim.x;
  for (; i < n; i += st) p[i] = v;
}

// f32 [R,C] -> bf16 transposed [C,R]; LDS-tiled, coalesced both sides. grid(C/32, R/32), 256 thr
__global__ __launch_bounds__(256) void cast_t_k(const float* __restrict__ in,
                                                bf16* __restrict__ out, int R, int Cc) {
  __shared__ __align__(16) bf16 tile[32][33];
  const int r0 = blockIdx.y * 32, c0 = blockIdx.x * 32;
  const int tx = threadIdx.x & 31, ty = threadIdx.x >> 5;
  for (int i = ty; i < 32; i += 8)
    tile[i][tx] = (bf16)in[(size_t)(r0 + i) * Cc + c0 + tx];
  __syncthreads();
  for (int i = ty; i < 32; i += 8)
    out[(size_t)(c0 + i) * R + r0 + tx] = tile[tx][i];
}

// per-head V transpose: qkv[S,3072] (v section) -> vT[(b*H+h)*64 + d][T]. grid(2,32,B*H)
__global__ __launch_bounds__(256) void vtrans_k(const bf16* __restrict__ qkv,
                                                bf16* __restrict__ vT) {
  __shared__ __align__(16) bf16 tile[32][33];
  const int bh = blockIdx.z;
  const int b = bh >> 4, h = bh & 15;
  const int t0 = blockIdx.y * 32;
  const int d0 = blockIdx.x * 32;
  const bf16* src = qkv + (size_t)b * kT * k3D + 2 * kD + h * kHD;
  const int tx = threadIdx.x & 31, ty = threadIdx.x >> 5;
  for (int i = ty; i < 32; i += 8)
    tile[i][tx] = src[(size_t)(t0 + i) * k3D + d0 + tx];
  __syncthreads();
  bf16* dst = vT + (size_t)bh * kHD * kT;
  for (int i = ty; i < 32; i += 8)
    dst[(size_t)(d0 + i) * kT + t0 + tx] = tile[tx][i];
}

// ---------------- WMMA GEMM (NT): C[M,N] = A[M,K] * B[N,K]^T (+bias) ----------------
// 128x128 block tile, 8 waves, each wave 64(M) x 32(N) = 4x2 fragments.
template <bool OUTB>
__global__ __launch_bounds__(256) void gemm_bf16_k(
    const bf16* __restrict__ A, int lda,
    const bf16* __restrict__ B, int ldb,
    const float* __restrict__ bias,
    float* __restrict__ Cf, bf16* __restrict__ Cb, int ldc, int K) {
  const int w  = threadIdx.x >> 5;
  const int m0 = blockIdx.y * 128 + (w >> 2) * 64;
  const int n0 = blockIdx.x * 128 + (w & 3) * 32;

  v8f acc[4][2] = {};

  for (int k0 = 0; k0 < K; k0 += 32) {
    if (k0 + 32 < K) {  // gfx1250 global_prefetch hint for next K tile of A
      __builtin_prefetch(A + (size_t)m0 * lda + k0 + 32, 0, 1);
    }
    v16bf aF[4];
#pragma unroll
    for (int i = 0; i < 4; ++i)
      aF[i] = ld_a16(A + (size_t)(m0 + i * 16) * lda + k0, lda);
    v16bf bF[2];
#pragma unroll
    for (int j = 0; j < 2; ++j)
      bF[j] = ld_b_nt(B + (size_t)(n0 + j * 16) * ldb + k0, ldb);
#pragma unroll
    for (int i = 0; i < 4; ++i)
#pragma unroll
      for (int j = 0; j < 2; ++j)
        acc[i][j] = WMMA_BF16(aF[i], bF[j], acc[i][j]);
  }

  const int lane = threadIdx.x & 31;
  const int nl = lane & 15;
  const int mh = (lane >> 4) << 3;
#pragma unroll
  for (int i = 0; i < 4; ++i)
#pragma unroll
    for (int j = 0; j < 2; ++j) {
      const int col = n0 + j * 16 + nl;
      const float bv = bias ? bias[col] : 0.0f;
#pragma unroll
      for (int r = 0; r < 8; ++r) {
        const int row = m0 + i * 16 + mh + r;
        const float v = acc[i][j][r] + bv;
        if (OUTB) Cb[(size_t)row * ldc + col] = (bf16)v;
        else      Cf[(size_t)row * ldc + col] = v;
      }
    }
}

// ---------------- flash attention ----------------
// grid = (T/64, B*H); 8 waves; Q tile 64 rows; key chunks of 128; online softmax.
// V is pre-transposed per head (vT[(bh*64+d)*T + t]) so P*V uses contiguous NT loads.
// K fragments are batch-loaded before the S WMMAs; V fragments are prefetched before
// the softmax barrier so their latency hides behind the scalar softmax phase.
__global__ __launch_bounds__(256) void attn_k(const bf16* __restrict__ qkv,
                                              const bf16* __restrict__ vT,
                                              bf16* __restrict__ ctx) {
  __shared__ __align__(16) float Sld[64 * 128];
  __shared__ __align__(16) bf16  Pld[64 * 128];
  __shared__ float rowM[64], rowL[64], rowC[64];

  const int w    = threadIdx.x >> 5;
  const int lane = threadIdx.x & 31;
  const int bh = blockIdx.y;
  const int b  = bh >> 4;
  const int h  = bh & 15;
  const int q0 = blockIdx.x * 64;

  const bf16* qp = qkv + (size_t)b * kT * k3D + h * kHD;
  const bf16* kp = qp + kD;
  const bf16* vp = vT + (size_t)bh * kHD * kT;

  const int mi = w >> 1;    // 0..3 : 16-row M fragment
  const int nb = w & 1;     // 0..1 : 64-key / 32-col half

  // Q fragments are invariant across key chunks
  v16bf aq0 = ld_a16(qp + (size_t)(q0 + mi * 16) * k3D + 0, k3D);
  v16bf aq1 = ld_a16(qp + (size_t)(q0 + mi * 16) * k3D + 32, k3D);

  v8f o0 = {}, o1 = {};
  if (threadIdx.x < 64) { rowM[threadIdx.x] = -3.0e38f; rowL[threadIdx.x] = 0.0f; }
  __syncthreads();

  for (int kc = 0; kc < kT; kc += 128) {
    // ---- batch-load all K fragments for this chunk ----
    v16bf bk[4][2];
#pragma unroll
    for (int j = 0; j < 4; ++j) {
      bk[j][0] = ld_b_nt(kp + (size_t)(kc + nb * 64 + j * 16) * k3D + 0, k3D);
      bk[j][1] = ld_b_nt(kp + (size_t)(kc + nb * 64 + j * 16) * k3D + 32, k3D);
    }
    // ---- S = Q * K^T ----
    v8f sf[4];
#pragma unroll
    for (int j = 0; j < 4; ++j) {
      v8f s = {};
      s = WMMA_BF16(aq0, bk[j][0], s);
      s = WMMA_BF16(aq1, bk[j][1], s);
      sf[j] = s;
    }
    // ---- scaled S -> LDS ----
    {
      const int mh = (lane >> 4) << 3;
#pragma unroll
      for (int j = 0; j < 4; ++j) {
        const int col = nb * 64 + j * 16 + (lane & 15);
#pragma unroll
        for (int r = 0; r < 8; ++r)
          Sld[(mi * 16 + mh + r) * 128 + col] = sf[j][r] * 0.125f;
      }
    }
    // ---- prefetch V fragments for this chunk (independent of softmax LDS work) ----
    v16bf vb[4][2];
#pragma unroll
    for (int ks = 0; ks < 4; ++ks) {
      vb[ks][0] = ld_b_nt(vp + (size_t)(nb * 32 + 0)  * kT + kc + ks * 32, kT);
      vb[ks][1] = ld_b_nt(vp + (size_t)(nb * 32 + 16) * kT + kc + ks * 32, kT);
    }
    __syncthreads();

    // ---- online softmax update (one thread per row) ----
    if (threadIdx.x < 64) {
      const int r = threadIdx.x;
      float mp = rowM[r], ml = mp;
      for (int c = 0; c < 128; ++c) ml = fmaxf(ml, Sld[r * 128 + c]);
      const float corr = __expf(mp - ml);
      float ls = rowL[r] * corr;
      for (int c = 0; c < 128; ++c) {
        const float pv = __expf(Sld[r * 128 + c] - ml);
        ls += pv;
        Pld[r * 128 + c] = (bf16)pv;
      }
      rowM[r] = ml; rowL[r] = ls; rowC[r] = corr;
    }
    __syncthreads();

    // ---- rescale accumulators ----
    {
      const int mh = (lane >> 4) << 3;
#pragma unroll
      for (int r = 0; r < 8; ++r) {
        const float c0 = rowC[mi * 16 + mh + r];
        o0[r] *= c0; o1[r] *= c0;
      }
    }
    // ---- ctx += P * V  (P from LDS, V fragments already in registers) ----
#pragma unroll
    for (int ks = 0; ks < 4; ++ks) {
      v16bf a = ld_a16(Pld + (size_t)(mi * 16) * 128 + ks * 32, 128);
      o0 = WMMA_BF16(a, vb[ks][0], o0);
      o1 = WMMA_BF16(a, vb[ks][1], o1);
    }
  }

  // ---- normalize + write ctx (bf16, [S, D] with head offset) ----
  const int mh = (lane >> 4) << 3;
  const int nl = lane & 15;
#pragma unroll
  for (int r = 0; r < 8; ++r) {
    const int row = mi * 16 + mh + r;
    const float inv = 1.0f / rowL[row];
    const size_t base = ((size_t)(b * kT + q0 + row)) * kD + h * kHD + nb * 32;
    ctx[base + nl]      = (bf16)(o0[r] * inv);
    ctx[base + 16 + nl] = (bf16)(o1[r] * inv);
  }
}

// ---------------- residual + layernorm (one block per row, D=1024) ----------------
__global__ __launch_bounds__(256) void add_ln_k(
    const float* __restrict__ a, const float* __restrict__ b,
    const float* __restrict__ g, const float* __restrict__ be,
    float* __restrict__ of, bf16* __restrict__ ob) {
  __shared__ float red[256];
  const int row = blockIdx.x;
  const int t = threadIdx.x;
  const float* pa = a + (size_t)row * kD;
  const float* pb = b ? b + (size_t)row * kD : nullptr;
  float v[4];
  float s = 0.0f;
#pragma unroll
  for (int i = 0; i < 4; ++i) {
    float x = pa[t * 4 + i] + (pb ? pb[t * 4 + i] : 0.0f);
    v[i] = x; s += x;
  }
  red[t] = s; __syncthreads();
  for (int st = 128; st > 0; st >>= 1) { if (t < st) red[t] += red[t + st]; __syncthreads(); }
  const float mean = red[0] * (1.0f / kD);
  __syncthreads();
  float s2 = 0.0f;
#pragma unroll
  for (int i = 0; i < 4; ++i) { const float d = v[i] - mean; s2 += d * d; }
  red[t] = s2; __syncthreads();
  for (int st = 128; st > 0; st >>= 1) { if (t < st) red[t] += red[t + st]; __syncthreads(); }
  const float rs = rsqrtf(red[0] * (1.0f / kD) + 1e-5f);
#pragma unroll
  for (int i = 0; i < 4; ++i) {
    const int c = t * 4 + i;
    const float y = (v[i] - mean) * rs * g[c] + be[c];
    if (of) of[(size_t)row * kD + c] = y;
    if (ob) ob[(size_t)row * kD + c] = (bf16)y;
  }
}

// ---------------- gate softmax + argmax (one block per token, E=4096) ----------------
__global__ __launch_bounds__(256) void gate_k(const float* __restrict__ logits,
                                              int* __restrict__ idx,
                                              float* __restrict__ gv) {
  __shared__ float rv[256];
  __shared__ int   ri[256];
  const int row = blockIdx.x;
  const int t = threadIdx.x;
  const float* p = logits + (size_t)row * kE;
  float lv[16];
  float bm = -3.0e38f; int bi = 0;
#pragma unroll
  for (int i = 0; i < 16; ++i) {
    const int c = t + i * 256;
    lv[i] = p[c];
    if (lv[i] > bm) { bm = lv[i]; bi = c; }   // strictly-greater keeps first index
  }
  rv[t] = bm; ri[t] = bi; __syncthreads();
  for (int st = 128; st > 0; st >>= 1) {
    if (t < st) {
      const float v2 = rv[t + st]; const int i2 = ri[t + st];
      if (v2 > rv[t] || (v2 == rv[t] && i2 < ri[t])) { rv[t] = v2; ri[t] = i2; }
    }
    __syncthreads();
  }
  const float mx = rv[0]; const int amax = ri[0];
  __syncthreads();
  float s = 0.0f;
#pragma unroll
  for (int i = 0; i < 16; ++i) s += __expf(lv[i] - mx);
  rv[t] = s; __syncthreads();
  for (int st = 128; st > 0; st >>= 1) { if (t < st) rv[t] += rv[t + st]; __syncthreads(); }
  if (t == 0) { idx[row] = amax; gv[row] = 1.0f / rv[0]; }  // softmax at argmax
}

// ---------------- capacity dispatch (deterministic token-order cumsum) ----------------
__global__ __launch_bounds__(1024) void dispatch_k(const int* __restrict__ idx,
                                                   float* __restrict__ gv,
                                                   int* __restrict__ slot) {
  __shared__ int cnt[kE];
  __shared__ int el[1024];
  const int t = threadIdx.x;
  for (int i = t; i < kE; i += 1024) cnt[i] = 0;
  __syncthreads();
  for (int ch = 0; ch < kS / 1024; ++ch) {
    const int s = ch * 1024 + t;
    const int e = idx[s];
    el[t] = e;
    __syncthreads();
    int rank = 0;
    for (int j = 0; j < t; ++j) rank += (el[j] == e);
    const int pos = cnt[e] + rank;
    if (pos < kC) slot[e * kC + pos] = s;
    else          gv[s] = 0.0f;               // dropped token
    __syncthreads();
    atomicAdd(&cnt[e], 1);
    __syncthreads();
  }
}

// ---------------- expert MLP (one block per expert; streams 128KB of weights) ----------
__global__ __launch_bounds__(256) void expert_k(
    const int* __restrict__ slot, const float* __restrict__ gv,
    const float* __restrict__ x1,
    const float* __restrict__ w1, const float* __restrict__ b1,
    const float* __restrict__ w2, const float* __restrict__ b2,
    float* __restrict__ moe) {
  __shared__ float hp[256][16];
  __shared__ float hs[kC][16];
  const int e = blockIdx.x;
  const int t = threadIdx.x;
  int tok[kC]; int nt = 0;
#pragma unroll
  for (int c = 0; c < kC; ++c) { tok[c] = slot[e * kC + c]; if (tok[c] >= 0) nt = c + 1; }
  if (nt == 0) return;

  const float* w1e = w1 + (size_t)e * kD * kNH;
  float xv[kC][4];
  for (int c = 0; c < nt; ++c) {
    const float* xr = x1 + (size_t)tok[c] * kD;
#pragma unroll
    for (int dd = 0; dd < 4; ++dd) xv[c][dd] = xr[t * 4 + dd];
  }
  float hacc[kC][16] = {};
#pragma unroll
  for (int dd = 0; dd < 4; ++dd) {
    const float* wr = w1e + (size_t)(t * 4 + dd) * kNH;
    float wreg[16];
#pragma unroll
    for (int j = 0; j < 16; ++j) wreg[j] = wr[j];
    for (int c = 0; c < nt; ++c)
#pragma unroll
      for (int j = 0; j < 16; ++j) hacc[c][j] += xv[c][dd] * wreg[j];
  }
  for (int c = 0; c < nt; ++c) {
#pragma unroll
    for (int j = 0; j < 16; ++j) hp[t][j] = hacc[c][j];
    __syncthreads();
    for (int st = 128; st > 0; st >>= 1) {
      if (t < st)
#pragma unroll
        for (int j = 0; j < 16; ++j) hp[t][j] += hp[t + st][j];
      __syncthreads();
    }
    if (t < 16) hs[c][t] = fmaxf(0.0f, hp[0][t] + b1[(size_t)e * kNH + t]);
    __syncthreads();
  }
  const float* w2e = w2 + (size_t)e * kNH * kD;
  float hr[kC][16];
  for (int c = 0; c < nt; ++c)
#pragma unroll
    for (int j = 0; j < 16; ++j) hr[c][j] = hs[c][j];
  float ya[kC][4] = {};
#pragma unroll
  for (int j = 0; j < 16; ++j)
#pragma unroll
    for (int dd = 0; dd < 4; ++dd) {
      const float wv = w2e[(size_t)j * kD + t + dd * 256];
      for (int c = 0; c < nt; ++c) ya[c][dd] += hr[c][j] * wv;
    }
  for (int c = 0; c < nt; ++c) {
    const float g = gv[tok[c]];
    float* orow = moe + (size_t)tok[c] * kD;
#pragma unroll
    for (int dd = 0; dd < 4; ++dd) {
      const int d = t + dd * 256;
      orow[d] = g * (ya[c][dd] + b2[(size_t)e * kD + d]);
    }
  }
}

// ---------------- launcher ----------------
extern "C" void kernel_launch(void* const* d_in, const int* in_sizes, int n_in,
                              void* d_out, int out_size, void* d_ws, size_t ws_size,
                              hipStream_t stream) {
  (void)in_sizes; (void)n_in; (void)out_size; (void)ws_size;
  const float* x    = (const float*)d_in[0];
  const float* Wqkv = (const float*)d_in[1];
  const float* bqkv = (const float*)d_in[2];
  const float* Wo   = (const float*)d_in[3];
  const float* bo   = (const float*)d_in[4];
  const float* g1   = (const float*)d_in[5];
  const float* be1  = (const float*)d_in[6];
  const float* g2   = (const float*)d_in[7];
  const float* be2  = (const float*)d_in[8];
  const float* Wg   = (const float*)d_in[9];
  const float* w1   = (const float*)d_in[10];
  const float* b1   = (const float*)d_in[11];
  const float* w2   = (const float*)d_in[12];
  const float* b2   = (const float*)d_in[13];

  char* p = (char*)d_ws;
  auto alloc = [&](size_t bytes) -> char* {
    char* r = p;
    p += (bytes + 255) & ~(size_t)255;
    return r;
  };
  bf16* xb     = (bf16*)alloc((size_t)kS * kD * 2);
  bf16* wqkvb  = (bf16*)alloc((size_t)k3D * kD * 2);
  bf16* woutb  = (bf16*)alloc((size_t)kD * kD * 2);
  bf16* wgT    = (bf16*)alloc((size_t)kE * kD * 2);   // gate_w transposed [E, D]
  bf16* qkvb   = (bf16*)alloc((size_t)kS * k3D * 2);
  bf16* vT     = (bf16*)alloc((size_t)kB * kH * kHD * kT * 2);  // per-head V^T
  bf16* ctxb   = (bf16*)alloc((size_t)kS * kD * 2);
  float* sa    = (float*)alloc((size_t)kS * kD * 4);
  float* x1    = (float*)alloc((size_t)kS * kD * 4);
  bf16* x1b    = (bf16*)alloc((size_t)kS * kD * 2);
  float* logit = (float*)alloc((size_t)kS * kE * 4);
  int*  idx    = (int*)alloc((size_t)kS * 4);
  float* gv    = (float*)alloc((size_t)kS * 4);
  int*  slot   = (int*)alloc((size_t)kE * kC * 4);
  float* moe   = (float*)alloc((size_t)kS * kD * 4);

  // bf16 copies of activations / weights (+ transposed gate weight)
  cast_k<<<4096, 256, 0, stream>>>(x,    xb,    (size_t)kS * kD);
  cast_k<<<2048, 256, 0, stream>>>(Wqkv, wqkvb, (size_t)k3D * kD);
  cast_k<<<1024, 256, 0, stream>>>(Wo,   woutb, (size_t)kD * kD);
  cast_t_k<<<dim3(kE / 32, kD / 32), 256, 0, stream>>>(Wg, wgT, kD, kE);

  // qkv = x @ Wqkv^T + b  (bf16 out)
  gemm_bf16_k<true><<<dim3(k3D / 128, kS / 128), 256, 0, stream>>>(
      xb, kD, wqkvb, kD, bqkv, (float*)nullptr, qkvb, k3D, kD);

  // per-head V transpose, then flash attention -> ctx (bf16)
  vtrans_k<<<dim3(kHD / 32, kT / 32, kB * kH), 256, 0, stream>>>(qkvb, vT);
  attn_k<<<dim3(kT / 64, kB * kH), 256, 0, stream>>>(qkvb, vT, ctxb);

  // sa = ctx @ Wo^T + bo (f32 out)
  gemm_bf16_k<false><<<dim3(kD / 128, kS / 128), 256, 0, stream>>>(
      ctxb, kD, woutb, kD, bo, sa, (bf16*)nullptr, kD, kD);

  // x1 = LN(x + sa)  (f32 + bf16)
  add_ln_k<<<kS, 256, 0, stream>>>(x, sa, g1, be1, x1, x1b);

  // logits = x1 @ gate_w = x1 @ wgT^T (NT, f32 out)
  gemm_bf16_k<false><<<dim3(kE / 128, kS / 128), 256, 0, stream>>>(
      x1b, kD, wgT, kD, (const float*)nullptr, logit, (bf16*)nullptr, kE, kD);

  // gate softmax/argmax, dispatch with capacity, expert MLP, combine
  gate_k<<<kS, 256, 0, stream>>>(logit, idx, gv);
  fill_i32_k<<<(kE * kC + 255) / 256, 256, 0, stream>>>(slot, -1, kE * kC);
  fill_f32_k<<<4096, 256, 0, stream>>>(moe, 0.0f, (size_t)kS * kD);
  dispatch_k<<<1, 1024, 0, stream>>>(idx, gv, slot);
  expert_k<<<kE, 256, 0, stream>>>(slot, gv, x1, w1, b1, w2, b2, moe);

  // out = LN(x1 + moe)
  add_ln_k<<<kS, 256, 0, stream>>>(x1, moe, g2, be2, (float*)d_out, (bf16*)nullptr);
}
